// VGG_Cifar10_classifieronly_3788161155090
// MI455X (gfx1250) — 8-bit matrix operand order repaired and checked by reading, compiled, not run on hardware
//
#include <hip/hip_runtime.h>
#include <stdint.h>


typedef int v8i __attribute__((ext_vector_type(8)));

#define BATCH 4096
#define DIN   8192
#define HID   1024
#define NCLS  1000
#define NCLSP 1024
#define EPSF  1e-5f
#define LDSTR 80

#if __has_builtin(__builtin_amdgcn_s_wait_asynccnt)
#define WAIT_ASYNCCNT(n) __builtin_amdgcn_s_wait_asynccnt(n)
#else
#define WAIT_ASYNCCNT(n) asm volatile("s_wait_asynccnt %0" :: "i"(n) : "memory")
#endif

__device__ __forceinline__ uint32_t lds_off(const void* p) {
  return (uint32_t)(uintptr_t)p;
}

__device__ __forceinline__ void async_b128(uint32_t lds, const int8_t* g) {
  asm volatile("global_load_async_to_lds_b128 %0, %1, off"
               :: "v"(lds), "v"(g) : "memory");
}

__global__ __launch_bounds__(256) void sign_pack(const float* __restrict__ in,
                                                 int8_t* __restrict__ out,
                                                 size_t n_total, size_t n_valid) {
  size_t stride = (size_t)gridDim.x * blockDim.x;
  for (size_t i = (size_t)blockIdx.x * blockDim.x + threadIdx.x; i < n_total; i += stride) {
    int8_t v = 0;
    if (i < n_valid) v = (in[i] >= 0.0f) ? (int8_t)1 : (int8_t)(-1);
    out[i] = v;
  }
}

__global__ void zero_u64(unsigned long long* __restrict__ p, int n) {
  int i = blockIdx.x * blockDim.x + threadIdx.x;
  if (i < n) p[i] = 0ull;
}

__global__ __launch_bounds__(256)
void bgemm_iu8_lds(const int8_t* __restrict__ A, const int8_t* __restrict__ W,
                   int* __restrict__ C, int M, int N, int K) {
  __shared__ int8_t sA[2][128 * LDSTR];
  __shared__ int8_t sB[2][128 * LDSTR];

  const int lane = threadIdx.x;
  const int wave = threadIdx.y;
  const int tid  = wave * 32 + lane;
  const int half = lane >> 4;
  const int sub  = lane & 15;
  const int wm   = wave & 3;
  const int wn   = wave >> 2;
  const int bm0  = blockIdx.y * 128;
  const int bn0  = blockIdx.x * 128;

  const int srow = tid >> 1;
  const int skof = (tid & 1) * 32;
  const int8_t* gA = A + (size_t)(bm0 + srow) * K + skof;
  const int8_t* gB = W + (size_t)(bn0 + srow) * K + skof;
  uint32_t lA[2], lB[2];
  lA[0] = lds_off(&sA[0][srow * LDSTR + skof]);
  lA[1] = lds_off(&sA[1][srow * LDSTR + skof]);
  lB[0] = lds_off(&sB[0][srow * LDSTR + skof]);
  lB[1] = lds_off(&sB[1][srow * LDSTR + skof]);

  v8i acc[8];
  const v8i vzero = {0, 0, 0, 0, 0, 0, 0, 0};
#pragma unroll
  for (int i = 0; i < 8; ++i) acc[i] = vzero;

  const int nsteps = K >> 6;

  async_b128(lA[0], gA);
  async_b128(lA[0] + 16, gA + 16);
  async_b128(lB[0], gB);
  async_b128(lB[0] + 16, gB + 16);

  int buf = 0;
  for (int ks = 0; ks < nsteps; ++ks) {
    if (ks + 1 < nsteps) {
      const int k1 = (ks + 1) << 6;
      async_b128(lA[buf ^ 1], gA + k1);
      async_b128(lA[buf ^ 1] + 16, gA + k1 + 16);
      async_b128(lB[buf ^ 1], gB + k1);
      async_b128(lB[buf ^ 1] + 16, gB + k1 + 16);
      WAIT_ASYNCCNT(4);
    } else {
      WAIT_ASYNCCNT(0);
    }
    __syncthreads();

    v8i afr[2];
#pragma unroll
    for (int mt = 0; mt < 2; ++mt) {
      const int8_t* ap = &sA[buf][(wm * 32 + mt * 16 + sub) * LDSTR + half * 8];
      int2 t0 = *(const int2*)(ap);
      int2 t1 = *(const int2*)(ap + 16);
      int2 t2 = *(const int2*)(ap + 32);
      int2 t3 = *(const int2*)(ap + 48);
      afr[mt][0] = t0.x; afr[mt][1] = t0.y; afr[mt][2] = t1.x; afr[mt][3] = t1.y;
      afr[mt][4] = t2.x; afr[mt][5] = t2.y; afr[mt][6] = t3.x; afr[mt][7] = t3.y;
    }
#pragma unroll
    for (int nt = 0; nt < 4; ++nt) {
      const int8_t* bp = &sB[buf][(wn * 64 + nt * 16 + sub) * LDSTR + half * 8];
      int2 u0 = *(const int2*)(bp);
      int2 u1 = *(const int2*)(bp + 16);
      int2 u2 = *(const int2*)(bp + 32);
      int2 u3 = *(const int2*)(bp + 48);
      v8i bfr;
      bfr[0] = u0.x; bfr[1] = u0.y; bfr[2] = u1.x; bfr[3] = u1.y;
      bfr[4] = u2.x; bfr[5] = u2.y; bfr[6] = u3.x; bfr[7] = u3.y;
      acc[nt]     = __builtin_amdgcn_wmma_i32_16x16x64_iu8(true, afr[0], true, bfr, acc[nt],     false, false);
      acc[4 + nt] = __builtin_amdgcn_wmma_i32_16x16x64_iu8(true, afr[1], true, bfr, acc[4 + nt], false, false);
    }
    __syncthreads();
    buf ^= 1;
  }

#pragma unroll
  for (int mt = 0; mt < 2; ++mt) {
#pragma unroll
    for (int nt = 0; nt < 4; ++nt) {
      int* crow = C + (size_t)(bm0 + wm * 32 + mt * 16 + half * 8) * N
                    + (bn0 + wn * 64 + nt * 16 + sub);
      v8i a = acc[mt * 4 + nt];
#pragma unroll
      for (int d = 0; d < 8; ++d) crow[(size_t)d * N] = a[d];
    }
  }
}

__global__ __launch_bounds__(256)
void col_stats(const int* __restrict__ H, int Mrows, int Nvalid, int stride,
               unsigned long long* __restrict__ csum,
               unsigned long long* __restrict__ csumsq, int rowsPerBlock) {
  int col = blockIdx.x * blockDim.x + threadIdx.x;
  if (col >= Nvalid) return;
  int r0 = blockIdx.y * rowsPerBlock;
  int r1 = r0 + rowsPerBlock; if (r1 > Mrows) r1 = Mrows;
  long long s = 0, s2 = 0;
  for (int r = r0; r < r1; ++r) {
    long long v = (long long)H[(size_t)r * stride + col];
    s += v; s2 += v * v;
  }
  atomicAdd(&csum[col],   (unsigned long long)s);
  atomicAdd(&csumsq[col], (unsigned long long)s2);
}

__global__ void finalize_stats(const unsigned long long* __restrict__ csum,
                               const unsigned long long* __restrict__ csumsq,
                               float* __restrict__ mean, float* __restrict__ invstd,
                               int Mrows, int Nvalid) {
  int c = blockIdx.x * blockDim.x + threadIdx.x;
  if (c >= Nvalid) return;
  double m = (double)(long long)csum[c] / (double)Mrows;
  double v = (double)(long long)csumsq[c] / (double)Mrows - m * m;
  mean[c]   = (float)m;
  invstd[c] = rsqrtf((float)v + EPSF);
}

__global__ __launch_bounds__(256)
void bn_sign(const int* __restrict__ H, const float* __restrict__ mean,
             const float* __restrict__ invstd, const float* __restrict__ g,
             const float* __restrict__ b, int8_t* __restrict__ out,
             size_t n, int Ncols) {
  size_t stride = (size_t)gridDim.x * blockDim.x;
  for (size_t i = (size_t)blockIdx.x * blockDim.x + threadIdx.x; i < n; i += stride) {
    int col = (int)(i & (size_t)(Ncols - 1));
    float y = ((float)H[i] - mean[col]) * invstd[col] * g[col] + b[col];
    out[i] = (y >= 0.0f) ? (int8_t)1 : (int8_t)(-1);
  }
}

__global__ __launch_bounds__(256)
void bn_logsoftmax(const int* __restrict__ H3, const float* __restrict__ mean,
                   const float* __restrict__ invstd, float* __restrict__ out,
                   float* __restrict__ hook) {
  __shared__ float ybuf[NCLS];
  __shared__ float red[256];
  int row = blockIdx.x;
  int tid = threadIdx.x;
  const int* hrow = H3 + (size_t)row * NCLSP;

  float lmax = -3.4e38f;
  for (int j = tid; j < NCLS; j += 256) {
    float v = ((float)hrow[j] - mean[j]) * invstd[j];
    ybuf[j] = v;
    lmax = fmaxf(lmax, v);
  }
  red[tid] = lmax; __syncthreads();
  for (int s = 128; s > 0; s >>= 1) {
    if (tid < s) red[tid] = fmaxf(red[tid], red[tid + s]);
    __syncthreads();
  }
  float mx = red[0]; __syncthreads();

  float lsum = 0.0f;
  for (int j = tid; j < NCLS; j += 256) lsum += expf(ybuf[j] - mx);
  red[tid] = lsum; __syncthreads();
  for (int s = 128; s > 0; s >>= 1) {
    if (tid < s) red[tid] += red[tid + s];
    __syncthreads();
  }
  float lse = mx + logf(red[0]);

  float* orow = out  + (size_t)row * NCLS;
  float* krow = hook + (size_t)row * NCLS;
  for (int j = tid; j < NCLS; j += 256) {
    float r = ybuf[j] - lse;
    orow[j] = r;
    krow[j] = r;
  }
}

extern "C" void kernel_launch(void* const* d_in, const int* in_sizes, int n_in,
                              void* d_out, int out_size, void* d_ws, size_t ws_size,
                              hipStream_t stream) {
  const float* x  = (const float*)d_in[0];
  const float* W1 = (const float*)d_in[1];
  const float* g1 = (const float*)d_in[2];
  const float* b1 = (const float*)d_in[3];
  const float* W2 = (const float*)d_in[4];
  const float* g2 = (const float*)d_in[5];
  const float* b2 = (const float*)d_in[6];
  const float* W3 = (const float*)d_in[7];
  float* out  = (float*)d_out;
  float* hook = out + (size_t)BATCH * NCLS;

  char* w = (char*)d_ws;
  auto alloc = [&](size_t bytes) -> char* {
    char* r = w; w += (bytes + 255) & ~(size_t)255; return r;
  };
  int8_t* Xb  = (int8_t*)alloc((size_t)BATCH * DIN);
  int8_t* W1b = (int8_t*)alloc((size_t)HID * DIN);
  int8_t* W2b = (int8_t*)alloc((size_t)HID * HID);
  int8_t* W3b = (int8_t*)alloc((size_t)NCLSP * HID);
  int*    Hbuf= (int*)   alloc((size_t)BATCH * HID * 4);
  int8_t* Xb2 = (int8_t*)alloc((size_t)BATCH * HID);
  unsigned long long* csum   = (unsigned long long*)alloc(HID * 8);
  unsigned long long* csumsq = (unsigned long long*)alloc(HID * 8);
  float* meanv = (float*)alloc(HID * 4);
  float* invv  = (float*)alloc(HID * 4);

  dim3 blk256(256);
  dim3 gblk(32, 8);
  dim3 ggrid(HID / 128, BATCH / 128);
  dim3 sgrid((HID + 255) / 256, 32);
  const int rowsPerBlock = BATCH / 32;

  sign_pack<<<4096, blk256, 0, stream>>>(x,  Xb,  (size_t)BATCH * DIN, (size_t)BATCH * DIN);
  sign_pack<<<2048, blk256, 0, stream>>>(W1, W1b, (size_t)HID * DIN,   (size_t)HID * DIN);
  sign_pack<<<512,  blk256, 0, stream>>>(W2, W2b, (size_t)HID * HID,   (size_t)HID * HID);
  sign_pack<<<512,  blk256, 0, stream>>>(W3, W3b, (size_t)NCLSP * HID, (size_t)NCLS * HID);

  bgemm_iu8_lds<<<ggrid, gblk, 0, stream>>>(Xb, W1b, Hbuf, BATCH, HID, DIN);
  zero_u64<<<8, blk256, 0, stream>>>(csum, HID);
  zero_u64<<<8, blk256, 0, stream>>>(csumsq, HID);
  col_stats<<<sgrid, blk256, 0, stream>>>(Hbuf, BATCH, HID, HID, csum, csumsq, rowsPerBlock);
  finalize_stats<<<4, blk256, 0, stream>>>(csum, csumsq, meanv, invv, BATCH, HID);
  bn_sign<<<2048, blk256, 0, stream>>>(Hbuf, meanv, invv, g1, b1, Xb2, (size_t)BATCH * HID, HID);

  bgemm_iu8_lds<<<ggrid, gblk, 0, stream>>>(Xb2, W2b, Hbuf, BATCH, HID, HID);
  zero_u64<<<8, blk256, 0, stream>>>(csum, HID);
  zero_u64<<<8, blk256, 0, stream>>>(csumsq, HID);
  col_stats<<<sgrid, blk256, 0, stream>>>(Hbuf, BATCH, HID, HID, csum, csumsq, rowsPerBlock);
  finalize_stats<<<4, blk256, 0, stream>>>(csum, csumsq, meanv, invv, BATCH, HID);
  bn_sign<<<2048, blk256, 0, stream>>>(Hbuf, meanv, invv, g2, b2, Xb2, (size_t)BATCH * HID, HID);

  bgemm_iu8_lds<<<ggrid, gblk, 0, stream>>>(Xb2, W3b, Hbuf, BATCH, NCLSP, HID);
  zero_u64<<<8, blk256, 0, stream>>>(csum, HID);
  zero_u64<<<8, blk256, 0, stream>>>(csumsq, HID);
  col_stats<<<sgrid, blk256, 0, stream>>>(Hbuf, BATCH, NCLS, NCLSP, csum, csumsq, rowsPerBlock);
  finalize_stats<<<4, blk256, 0, stream>>>(csum, csumsq, meanv, invv, BATCH, NCLS);

  bn_logsoftmax<<<BATCH, blk256, 0, stream>>>(Hbuf, meanv, invv, out, hook);
}
